// MultiScaleDecoderRetention_66700842107239
// MI455X (gfx1250) — matrix operand order repaired and checked by reading, not compiled, not run on hardware
//
#include <hip/hip_runtime.h>
#include <math.h>

typedef _Float16 v8h  __attribute__((ext_vector_type(8)));
typedef _Float16 v16h __attribute__((ext_vector_type(16)));
typedef float    v8f  __attribute__((ext_vector_type(8)));

#define HIDDEN 1024
#define HEADS  16
#define HEAD   64
#define B_SZ   2
#define LQ     2048
#define LK     2048

static __device__ __forceinline__ v16h cat8(v8h lo, v8h hi) {
  return __builtin_shufflevector(lo, hi, 0,1,2,3,4,5,6,7,8,9,10,11,12,13,14,15);
}

// ---------------------------------------------------------------------------
// 1. f32 -> f16 elementwise convert
// ---------------------------------------------------------------------------
__global__ void cvt_f16_kernel(const float* __restrict__ in,
                               _Float16* __restrict__ out, int n) {
  int i = blockIdx.x * blockDim.x + threadIdx.x;
  if (i < n) out[i] = (_Float16)in[i];
}

// ---------------------------------------------------------------------------
// 2. Pack a KxN f32 matrix into a WMMA 16-bit B-fragment layout (contiguous k order):
//    Bp[((kblk*nBlk + nblk)*32 + lane)*16 + e] =
//        W[(kblk*32 + (lane>=16)*16 + e)*N + nblk*16 + (lane&15)]
//    (k order chosen here, matched by the A loads: 16 consecutive K for a fixed column, lane-half
//     selects which 16 of the 32 K values.)
// ---------------------------------------------------------------------------
__global__ void pack_b_kernel(const float* __restrict__ W,
                              _Float16* __restrict__ Bp,
                              int K, int N, long wZStride, long bZStride) {
  W  += (long)blockIdx.z * wZStride;
  Bp += (long)blockIdx.z * bZStride;
  long t = (long)blockIdx.x * blockDim.x + threadIdx.x;
  long total = (long)K * N;
  if (t >= total) return;
  int  e    = (int)(t & 15);
  int  lane = (int)((t >> 4) & 31);
  long blk  = t >> 9;                 // /(16*32)
  int  nBlk = N >> 4;
  int  nblk = (int)(blk % nBlk);
  int  kblk = (int)(blk / nBlk);
  int  k = kblk * 32 + ((lane >= 16) ? 16 : 0) + e;
  int  n = nblk * 16 + (lane & 15);
  Bp[t] = (_Float16)W[(long)k * N + n];
}

// ---------------------------------------------------------------------------
// 3. Generic WMMA GEMM: C(f32, MxN row-major) = A(f16 row-major) * Bp(packed)
//    128 threads = 4 waves, each wave owns a 16x64 tile; K-step 32.
// ---------------------------------------------------------------------------
__global__ __launch_bounds__(128) void gemm16_kernel(
    const _Float16* __restrict__ A, const _Float16* __restrict__ Bp,
    float* __restrict__ C, int K, int lda, int ldc, int nBlkB,
    int headsMod, long aZStride, long bZStride, long cZStride) {
  int z = blockIdx.z;
  A  += (long)(z / headsMod) * aZStride;
  Bp += (long)(z % headsMod) * bZStride;
  C  += (long)z * cZStride;

  int lane = threadIdx.x & 31;
  int wave = threadIdx.x >> 5;
  int hi   = lane >> 4;          // lane-half selects K sub-range
  int m15  = lane & 15;
  int mtile = blockIdx.y * 64 + wave * 16;
  const _Float16* arow = A + (long)(mtile + m15) * lda;

  v8f z8 = {0.f,0.f,0.f,0.f,0.f,0.f,0.f,0.f};
  v8f acc0 = z8, acc1 = z8, acc2 = z8, acc3 = z8;

  for (int kk = 0; kk < K; kk += 32) {
    int kblk = kk >> 5;
    // A fragment (16x32 f16): k = kk + hi*16 + e (contiguous, matches packed B)
    v8h lo = *(const v8h*)(arow + kk + hi * 16);
    v8h hh = *(const v8h*)(arow + kk + 8 + hi * 16);
    v16h a = cat8(lo, hh);
    const _Float16* bbase =
        Bp + (((long)kblk * nBlkB + (long)blockIdx.x * 4) * 32 + lane) * 16;
    v16h b0 = *(const v16h*)(bbase);
    v16h b1 = *(const v16h*)(bbase + 512);
    v16h b2 = *(const v16h*)(bbase + 1024);
    v16h b3 = *(const v16h*)(bbase + 1536);
    if (kk + 32 < K) __builtin_prefetch(arow + kk + 32, 0, 1);
    acc0 = __builtin_amdgcn_wmma_f32_16x16x32_f16(false, a, false, b0, (short)0, acc0, false, false);
    acc1 = __builtin_amdgcn_wmma_f32_16x16x32_f16(false, a, false, b1, (short)0, acc1, false, false);
    acc2 = __builtin_amdgcn_wmma_f32_16x16x32_f16(false, a, false, b2, (short)0, acc2, false, false);
    acc3 = __builtin_amdgcn_wmma_f32_16x16x32_f16(false, a, false, b3, (short)0, acc3, false, false);
  }

  int ntile = blockIdx.x * 64;
  #pragma unroll
  for (int r = 0; r < 8; r++) {
    // C/D layout: VGPR r -> row mtile + r + hi*8, col = ntile + m15 (+16*j)
    float* crow = C + (long)(mtile + r + hi * 8) * ldc + ntile + m15;
    crow[0]  = acc0[r];
    crow[16] = acc1[r];
    crow[32] = acc2[r];
    crow[48] = acc3[r];
  }
}

// ---------------------------------------------------------------------------
// 4. xPos rotary scaling; input f32 (z=b*H+h, L, 64), output f16 row-major.
// ---------------------------------------------------------------------------
__global__ void xpos_kernel(const float* __restrict__ in,
                            _Float16* __restrict__ out, int downscale) {
  long t = (long)blockIdx.x * blockDim.x + threadIdx.x;
  int  i2 = (int)(t & 31);
  long r  = t >> 5;
  int  l  = (int)(r & (LQ - 1));
  long zz = r >> 11;
  if (zz >= (long)B_SZ * HEADS) return;
  float sv    = (2.0f * i2 + 0.4f * 64.0f) / (1.4f * 64.0f);
  float scale = powf(sv, (float)l / 512.0f);
  if (downscale) scale = 1.0f / scale;
  float invf = powf(10000.0f, -(float)i2 / 32.0f);
  float ang  = (float)l * invf;
  float ss = sinf(ang) * scale;
  float cs = cosf(ang) * scale;
  long base = (zz * LQ + l) * 64 + 2 * i2;
  float x1 = in[base], x2 = in[base + 1];
  out[base]     = (_Float16)(x1 * cs - x2 * ss);
  out[base + 1] = (_Float16)(x2 * cs + x1 * ss);
}

// ---------------------------------------------------------------------------
// 5. Retention attention: scores = (Q Kᵀ) * gamma^(i-j) (causal), Ret = S V.
//    One wave = one 16-query strip. Kh row-major acts as Kᵀ B-fragments.
//    Score f16 tiles go through LDS to be re-fetched as A fragments.
// ---------------------------------------------------------------------------
__global__ __launch_bounds__(128) void retention_attn_kernel(
    const _Float16* __restrict__ Qh, const _Float16* __restrict__ Kh,
    const _Float16* __restrict__ Vp, float* __restrict__ Ret) {
  int h = blockIdx.y;
  long zz = (long)blockIdx.z * HEADS + h;
  Qh  += zz * (long)LQ * 64;
  Kh  += zz * (long)LK * 64;
  Vp  += zz * (long)LK * 64;
  Ret += zz * (long)LQ * 64;

  const float xmin = -3.4657359028f;   // log(1/32)
  const float xmax = -6.2383246250f;   // log(1/512)
  float gamma = 1.0f - expf(xmin + (xmax - xmin) * ((float)h / 15.0f));
  float l2g   = log2f(gamma);

  int lane = threadIdx.x & 31;
  int wave = threadIdx.x >> 5;
  int hi   = lane >> 4;
  int m15  = lane & 15;
  int qt   = blockIdx.x * 64 + wave * 16;

  __shared__ _Float16 sS[4][16 * 32];
  _Float16* my = &sS[wave][0];

  const _Float16* qrow = Qh + (long)(qt + m15) * 64;
  v16h aq0 = cat8(*(const v8h*)(qrow +      hi * 16), *(const v8h*)(qrow + 8 + hi * 16));
  v16h aq1 = cat8(*(const v8h*)(qrow + 32 + hi * 16), *(const v8h*)(qrow + 40 + hi * 16));

  v8f z8 = {0.f,0.f,0.f,0.f,0.f,0.f,0.f,0.f};
  v8f acc0 = z8, acc1 = z8, acc2 = z8, acc3 = z8;

  int kbMax = (qt + 15) >> 5;
  for (int kb = 0; kb <= kbMax; kb++) {
    int keyBase = kb * 32;
    if (kb < kbMax)
      __builtin_prefetch(Kh + (long)(keyBase + 32 + m15) * 64, 0, 1);
    #pragma unroll
    for (int t = 0; t < 2; t++) {
      int key = keyBase + t * 16 + m15;
      const _Float16* krow = Kh + (long)key * 64;
      // B fragment of Kᵀ: 16 consecutive head-dims for a fixed key column.
      v16h bk0 = *(const v16h*)(krow + hi * 16);
      v16h bk1 = *(const v16h*)(krow + 32 + hi * 16);
      v8f s = z8;
      s = __builtin_amdgcn_wmma_f32_16x16x32_f16(false, aq0, false, bk0, (short)0, s, false, false);
      s = __builtin_amdgcn_wmma_f32_16x16x32_f16(false, aq1, false, bk1, (short)0, s, false, false);
      #pragma unroll
      for (int r = 0; r < 8; r++) {
        int i = qt + r + hi * 8;
        int j = keyBase + t * 16 + m15;
        float d  = (j <= i) ? exp2f(l2g * (float)(i - j)) : 0.0f;
        my[(r + hi * 8) * 32 + t * 16 + m15] = (_Float16)(s[r] * d);
      }
    }
    asm volatile("s_wait_dscnt 0" ::: "memory");  // scores visible in LDS
    v16h as = cat8(*(const v8h*)(my + m15 * 32 + hi * 16),
                   *(const v8h*)(my + m15 * 32 + 8 + hi * 16));
    asm volatile("" ::: "memory");
    const _Float16* vb = Vp + ((long)kb * 4 * 32 + lane) * 16;
    v16h b0 = *(const v16h*)(vb);
    v16h b1 = *(const v16h*)(vb + 512);
    v16h b2 = *(const v16h*)(vb + 1024);
    v16h b3 = *(const v16h*)(vb + 1536);
    acc0 = __builtin_amdgcn_wmma_f32_16x16x32_f16(false, as, false, b0, (short)0, acc0, false, false);
    acc1 = __builtin_amdgcn_wmma_f32_16x16x32_f16(false, as, false, b1, (short)0, acc1, false, false);
    acc2 = __builtin_amdgcn_wmma_f32_16x16x32_f16(false, as, false, b2, (short)0, acc2, false, false);
    acc3 = __builtin_amdgcn_wmma_f32_16x16x32_f16(false, as, false, b3, (short)0, acc3, false, false);
  }

  #pragma unroll
  for (int r = 0; r < 8; r++) {
    float* orow = Ret + (long)(qt + r + hi * 8) * 64 + m15;
    orow[0]  = acc0[r];
    orow[16] = acc1[r];
    orow[32] = acc2[r];
    orow[48] = acc3[r];
  }
}

// ---------------------------------------------------------------------------
// 6. GroupNorm(64 per head) + affine + SiLU(gate) -> f16 U
// ---------------------------------------------------------------------------
__global__ void groupnorm_gate_kernel(const float* __restrict__ Ret,
                                      const float* __restrict__ Gate,
                                      const float* __restrict__ gw,
                                      const float* __restrict__ gb,
                                      _Float16* __restrict__ U) {
  int t = blockIdx.x * blockDim.x + threadIdx.x;
  if (t >= B_SZ * LQ * HEADS) return;
  int h = t & 15;
  int l = (t >> 4) & (LQ - 1);
  int b = t >> 15;
  const float* r = Ret + (((long)(b * HEADS + h)) * LQ + l) * 64;
  float sum = 0.f, sq = 0.f;
  for (int c = 0; c < 64; c++) { float x = r[c]; sum += x; sq += x * x; }
  float mean = sum * (1.0f / 64.0f);
  float var  = sq * (1.0f / 64.0f) - mean * mean;
  float inv  = rsqrtf(var + 1e-5f);
  long obase = ((long)(b * LQ + l)) * HIDDEN + h * 64;
  for (int c = 0; c < 64; c++) {
    float rn = (r[c] - mean) * inv;
    float y  = rn * gw[h * 64 + c] + gb[h * 64 + c];
    float g  = Gate[obase + c];
    float sg = g / (1.0f + expf(-g));
    U[obase + c] = (_Float16)(sg * y);
  }
}

// ---------------------------------------------------------------------------
extern "C" void kernel_launch(void* const* d_in, const int* in_sizes, int n_in,
                              void* d_out, int out_size, void* d_ws, size_t ws_size,
                              hipStream_t stream) {
  const float* X   = (const float*)d_in[0];
  const float* Mem = (const float*)d_in[1];
  const float* W_Q = (const float*)d_in[2];
  const float* W_K = (const float*)d_in[3];
  const float* W_V = (const float*)d_in[4];
  const float* W_G = (const float*)d_in[5];
  const float* W_O = (const float*)d_in[6];
  const float* gw  = (const float*)d_in[7];
  const float* gb  = (const float*)d_in[8];
  float* out = (float*)d_out;

  char* ws = (char*)d_ws;
  const long nX = (long)B_SZ * LQ * HIDDEN;  // 4,194,304

  _Float16* Xh  = (_Float16*)(ws + 0);
  _Float16* Mh  = (_Float16*)(ws + 8388608);
  _Float16* WQp = (_Float16*)(ws + 16777216);
  _Float16* WKp = (_Float16*)(ws + 18874368);
  _Float16* WVp = (_Float16*)(ws + 20971520);
  _Float16* WGp = (_Float16*)(ws + 23068672);
  _Float16* WOp = (_Float16*)(ws + 25165824);
  float*    Q32 = (float*)   (ws + 27262976);  // later reused as Gate
  float*    K32 = (float*)   (ws + 44040192);  // later reused as Ret
  float*    V32 = (float*)   (ws + 60817408);  // later reused as U (f16)
  _Float16* Qh  = (_Float16*)(ws + 77594624);
  _Float16* Kh  = (_Float16*)(ws + 85983232);
  _Float16* Vp  = (_Float16*)(ws + 94371840);
  float*    Gate = Q32;
  float*    Ret  = K32;
  _Float16* U    = (_Float16*)V32;

  dim3 blk(256);

  // 1. f32 -> f16
  cvt_f16_kernel<<<dim3((unsigned)((nX + 255) / 256)), blk, 0, stream>>>(X, Xh, (int)nX);
  cvt_f16_kernel<<<dim3((unsigned)((nX + 255) / 256)), blk, 0, stream>>>(Mem, Mh, (int)nX);

  // 2. pack weights into B-fragment layout
  long whStride = (long)HIDDEN * HEAD;  // 65536 per head
  pack_b_kernel<<<dim3(256, 1, HEADS), blk, 0, stream>>>(W_Q, WQp, HIDDEN, HEAD, whStride, whStride);
  pack_b_kernel<<<dim3(256, 1, HEADS), blk, 0, stream>>>(W_K, WKp, HIDDEN, HEAD, whStride, whStride);
  pack_b_kernel<<<dim3(256, 1, HEADS), blk, 0, stream>>>(W_V, WVp, HIDDEN, HEAD, whStride, whStride);
  pack_b_kernel<<<dim3(4096, 1, 1), blk, 0, stream>>>(W_G, WGp, HIDDEN, HIDDEN, 0, 0);
  pack_b_kernel<<<dim3(4096, 1, 1), blk, 0, stream>>>(W_O, WOp, HIDDEN, HIDDEN, 0, 0);

  // 3. Q/K/V projections (z = b*HEADS + h)
  dim3 gProj(1, LQ / 64, B_SZ * HEADS);
  gemm16_kernel<<<gProj, dim3(128), 0, stream>>>(Xh, WQp, Q32, HIDDEN, HIDDEN, HEAD, HEAD / 16,
      HEADS, (long)LQ * HIDDEN, whStride, (long)LQ * HEAD);
  gemm16_kernel<<<gProj, dim3(128), 0, stream>>>(Mh, WKp, K32, HIDDEN, HIDDEN, HEAD, HEAD / 16,
      HEADS, (long)LK * HIDDEN, whStride, (long)LK * HEAD);
  gemm16_kernel<<<gProj, dim3(128), 0, stream>>>(Mh, WVp, V32, HIDDEN, HIDDEN, HEAD, HEAD / 16,
      HEADS, (long)LK * HIDDEN, whStride, (long)LK * HEAD);

  // 4. xPos (Q up-scale, K down-scale) -> f16
  xpos_kernel<<<dim3(8192), blk, 0, stream>>>(Q32, Qh, 0);
  xpos_kernel<<<dim3(8192), blk, 0, stream>>>(K32, Kh, 1);

  // 5. pack V into B-fragment layout (per b,h)
  pack_b_kernel<<<dim3(512, 1, B_SZ * HEADS), blk, 0, stream>>>(V32, Vp, LK, HEAD,
      (long)LK * HEAD, (long)LK * HEAD);

  // 6. retention attention
  retention_attn_kernel<<<dim3(LQ / 64, HEADS, B_SZ), dim3(128), 0, stream>>>(Qh, Kh, Vp, Ret);

  // 7. gate = X @ W_G (silu applied in combine)
  dim3 gBig(HIDDEN / 64, (B_SZ * LQ) / 64, 1);
  gemm16_kernel<<<gBig, dim3(128), 0, stream>>>(Xh, WGp, Gate, HIDDEN, HIDDEN, HIDDEN, HIDDEN / 16,
      1, 0, 0, 0);

  // 8. groupnorm + gate -> U (f16)
  groupnorm_gate_kernel<<<dim3((B_SZ * LQ * HEADS) / 256), blk, 0, stream>>>(Ret, Gate, gw, gb, U);

  // 9. out = U @ W_O
  gemm16_kernel<<<gBig, dim3(128), 0, stream>>>(U, WOp, out, HIDDEN, HIDDEN, HIDDEN, HIDDEN / 16,
      1, 0, 0, 0);
}
